// GRU_28948079575637
// MI455X (gfx1250) — matrix operand order repaired and checked by reading, not compiled, not run on hardware
//
#include <hip/hip_runtime.h>
#include <hip/hip_bf16.h>
#include <stdint.h>

#define T_STEPS 512
#define BATCH   64
#define EMB     512
#define RNN     1024
#define OUTN    4
#define NWG     64        // persistent workgroups; each owns RNN/NWG = 16 hidden units
#define JW      16        // hidden units per workgroup
#define THREADS 384       // 12 waves = 4 batch tiles x 3 gates

typedef __attribute__((ext_vector_type(16))) __bf16 v16bf;
typedef __attribute__((ext_vector_type(8)))  float  v8f;
typedef __attribute__((ext_vector_type(4)))  unsigned int v4ui;
typedef __attribute__((ext_vector_type(8)))  int    v8si;
typedef __attribute__((ext_vector_type(4)))  int    v4si;

#ifndef __has_builtin
#define __has_builtin(x) 0
#endif
#if defined(__HIP_DEVICE_COMPILE__) && __has_builtin(__builtin_amdgcn_tensor_load_to_lds)
#define USE_TDM 1
#else
#define USE_TDM 0
#endif

// LDS stream padding: +8 bf16 (4 dwords, 16B) after every 512 bf16 (256 dwords).
// Matches TDM pad_interval=256dw / pad_amount=4dw; keeps 16B alignment for b128 LDS loads.
__device__ __forceinline__ uint32_t pad_lds(uint32_t e) { return e + (e >> 9) * 8u; }

// padded element counts for LDS slices
#define WIH_ELE (3u * JW * EMB)                       // 24576
#define WHH_ELE (3u * JW * RNN)                       // 49152
#define WIH_LDS (WIH_ELE + (WIH_ELE >> 9) * 8u)       // 24960
#define WHH_LDS (WHH_ELE + (WHH_ELE >> 9) * 8u)       // 49920

// ---- WMMA operand loaders (wave32; contiguous k order, same in A and B) ----
// A 16x32 bf16: lane L -> row m=L%16; elems 0..15 = K[kh..kh+15],
// kh = 16*(L>=16). Two 16B loads from a row-major bf16 row.
__device__ __forceinline__ v16bf load_a16(const __bf16* rowk0, int kh) {
  union { uint4 u[2]; v16bf v; } ua;
  ua.u[0] = *(const uint4*)(rowk0 + kh);
  ua.u[1] = *(const uint4*)(rowk0 + kh + 8);
  return ua.v;
}
// B 32x16 bf16: lane L -> col n=L%16; elems 0..15 = K[kh16 .. kh16+15], kh16 = 16*(L>=16).
// = 32 contiguous bytes of one weight row in the padded LDS stream.
__device__ __forceinline__ v16bf load_b16(const __bf16* s, uint32_t stream_elem) {
  const uint32_t e = pad_lds(stream_elem);
  union { uint4 u[2]; v16bf v; } ub;
  ub.u[0] = *(const uint4*)(s + e);
  ub.u[1] = *(const uint4*)(s + e + 8);
  return ub.v;
}

__device__ __forceinline__ v8f wmma_bf16(v16bf a, v16bf b, v8f c) {
  // 8 args: (neg_a, A, neg_b, B, c_mod, C, reuse_a, reuse_b)
  return __builtin_amdgcn_wmma_f32_16x16x32_bf16(false, a, false, b, (short)0, c, false, false);
}

__device__ __forceinline__ float sigmoidf(float x) { return 1.0f / (1.0f + __expf(-x)); }

#if USE_TDM
// Issue one TENSOR_LOAD_TO_LDS for a 3D tile: [ngate][nrows][row_len] bf16,
// row stride = row_len, gate stride = gate_stride_elems, with LDS padding enabled.
__device__ __forceinline__ void tdm_load_3d(uint32_t lds_base, const __bf16* gsrc,
                                            uint32_t row_len, uint32_t nrows,
                                            uint32_t ngate, uint64_t gate_stride_elems) {
  const uint64_t ga = (uint64_t)(uintptr_t)gsrc;
  v4ui g0;
  g0[0] = 1u;                                   // count=1 user descriptor
  g0[1] = lds_base;                             // lds_addr (bytes, LDS offset)
  g0[2] = (uint32_t)ga;                         // global_addr[31:0]
  g0[3] = ((uint32_t)(ga >> 32) & 0x01FFFFFFu) | (2u << 30);  // addr[56:32] | type=2
  const uint32_t dim1 = 1u << 20;               // generous OOB bound
  v8si g1;
  g1[0] = (int)((1u << 16)      // data_size = 2B
              | (1u << 20)      // pad_enable
              | (7u << 22)      // pad_interval = 256 dwords
              | (3u << 25));    // pad_amount  = 4 dwords
  g1[1] = (int)((row_len & 0xFFFFu) << 16);                       // tensor_dim0 lo16
  g1[2] = (int)((row_len >> 16) | ((dim1 & 0xFFFFu) << 16));      // dim0 hi16 | dim1 lo16
  g1[3] = (int)((dim1 >> 16) | ((row_len & 0xFFFFu) << 16));      // dim1 hi16 | tile_dim0
  g1[4] = (int)((nrows & 0xFFFFu) | ((ngate & 0xFFFFu) << 16));   // tile_dim1 | tile_dim2
  const uint64_t s0 = row_len, s1 = gate_stride_elems;
  g1[5] = (int)(uint32_t)s0;                                      // dim0_stride lo32
  g1[6] = (int)(((uint32_t)(s0 >> 32) & 0xFFFFu) | (((uint32_t)s1 & 0xFFFFu) << 16));
  g1[7] = (int)(uint32_t)(s1 >> 16);                              // dim1_stride hi32
  v4si g2; g2[0] = (int)ngate; g2[1] = 0; g2[2] = 0; g2[3] = 0;   // tensor_dim2
  v4si g3 = {0, 0, 0, 0};
#if __clang_major__ >= 23
  v8si gz = {0, 0, 0, 0, 0, 0, 0, 0};
  __builtin_amdgcn_tensor_load_to_lds(g0, g1, g2, g3, gz, 0);
#else
  __builtin_amdgcn_tensor_load_to_lds(g0, g1, g2, g3, 0);
#endif
}
#endif

// ------------------- phase kernels -------------------
__global__ void k_cvt(const float* __restrict__ s, __bf16* __restrict__ d, int n) {
  int i = blockIdx.x * blockDim.x + threadIdx.x;
  if (i < n) d[i] = (__bf16)s[i];
}

__global__ void k_embed(const int* __restrict__ x, const float* __restrict__ tab,
                        __bf16* __restrict__ emb) {
  const int r = blockIdx.x;                 // 0 .. T*B-1
  const int idx = x[r];
  const float* src = tab + (size_t)idx * EMB;
  __bf16* dst = emb + (size_t)r * EMB;
  for (int i = threadIdx.x; i < EMB; i += blockDim.x) dst[i] = (__bf16)src[i];
}

__global__ void k_init(float* __restrict__ hf, __bf16* __restrict__ hb,
                       unsigned* __restrict__ ctr) {
  const int stride = gridDim.x * blockDim.x;
  for (int e = blockIdx.x * blockDim.x + threadIdx.x; e < 2 * BATCH * RNN; e += stride) {
    hf[e] = 0.0f;
    hb[e] = (__bf16)0.0f;
  }
  if (blockIdx.x == 0 && threadIdx.x == 0) *ctr = 0u;
}

// ------------------- persistent GRU kernel -------------------
__global__ __launch_bounds__(THREADS, 1)
void k_gru(const __bf16* __restrict__ wih,  // [3*RNN][EMB] bf16
           const __bf16* __restrict__ whh,  // [3*RNN][RNN] bf16
           const __bf16* __restrict__ embb, // [T][B][EMB] bf16
           const float* __restrict__ bih, const float* __restrict__ bhh,
           float*  __restrict__ hf,         // [2][B][RNN] f32
           __bf16* __restrict__ hb,         // [2][B][RNN] bf16
           unsigned* __restrict__ ctr) {
  __shared__ __bf16 s_wih[WIH_LDS];
  __shared__ __bf16 s_whh[WHH_LDS];
  __shared__ float  s_acc[4][BATCH][JW];    // 0:r_pre 1:z_pre 2:gi_n 3:gh_n

  const int tid   = threadIdx.x;
  const int lane  = tid & 31;
  const int wave  = tid >> 5;               // 0..11
  const int mtile = wave & 3;               // batch tile (16 rows)
  const int role  = wave >> 2;              // 0=r 1=z 2=n
  const int j0    = blockIdx.x * JW;        // hidden-unit slice

  // ---- load this WG's weight slices into LDS (resident for all 512 steps) ----
#if USE_TDM
  if (wave == 0) {
    tdm_load_3d((uint32_t)(uintptr_t)(void*)s_wih, wih + (size_t)j0 * EMB,
                EMB, JW, 3, (uint64_t)RNN * EMB);
    tdm_load_3d((uint32_t)(uintptr_t)(void*)s_whh, whh + (size_t)j0 * RNN,
                RNN, JW, 3, (uint64_t)RNN * RNN);
    __builtin_amdgcn_s_wait_tensorcnt(0);
  }
#else
  for (uint32_t q = tid; q < (WIH_ELE >> 2); q += THREADS) {  // 4 bf16 per copy
    const uint32_t e = q << 2;
    const uint32_t row = e >> 9, k = e & (EMB - 1);           // EMB = 512
    const uint32_t gate = row >> 4, jj = row & 15u;
    *(uint64_t*)(s_wih + pad_lds(e)) =
        *(const uint64_t*)(wih + ((size_t)(gate * RNN + j0 + jj) * EMB + k));
  }
  for (uint32_t q = tid; q < (WHH_ELE >> 2); q += THREADS) {
    const uint32_t e = q << 2;
    const uint32_t row = e >> 10, k = e & (RNN - 1);          // RNN = 1024
    const uint32_t gate = row >> 4, jj = row & 15u;
    *(uint64_t*)(s_whh + pad_lds(e)) =
        *(const uint64_t*)(whh + ((size_t)(gate * RNN + j0 + jj) * RNN + k));
  }
#endif
  __syncthreads();

  const int a_m  = lane & 15;               // A-row within tile
  const int a_kh = (lane >> 4) << 4;        // A k sub-offset (0/16)
  const int b_n  = lane & 15;               // B-col
  const int b_kh = (lane >> 4) << 4;        // B k sub-offset (0/16)

  for (int t = 0; t < T_STEPS; ++t) {
    const int p = t & 1;
    const __bf16* hcur = hb + (size_t)p * (BATCH * RNN);
    const __bf16* arow_e = embb + ((size_t)t * BATCH + mtile * 16 + a_m) * EMB;
    const __bf16* arow_h = hcur + (size_t)(mtile * 16 + a_m) * RNN;

    v8f acc  = {0.f, 0.f, 0.f, 0.f, 0.f, 0.f, 0.f, 0.f};  // r/z: gi+gh ; n: gh
    v8f acc2 = {0.f, 0.f, 0.f, 0.f, 0.f, 0.f, 0.f, 0.f};  // n only: gi

    // gi part: A = embedded[t], B = w_ih slice, K = EMB
    if (role < 2) {
#pragma unroll 4
      for (int k0 = 0; k0 < EMB; k0 += 32)
        acc = wmma_bf16(load_a16(arow_e + k0, a_kh),
                        load_b16(s_wih, (uint32_t)(role * JW + b_n) * EMB + k0 + b_kh), acc);
    } else {
#pragma unroll 4
      for (int k0 = 0; k0 < EMB; k0 += 32)
        acc2 = wmma_bf16(load_a16(arow_e + k0, a_kh),
                         load_b16(s_wih, (uint32_t)(2 * JW + b_n) * EMB + k0 + b_kh), acc2);
    }
    // gh part: A = h[t], B = w_hh slice, K = RNN
#pragma unroll 4
    for (int k0 = 0; k0 < RNN; k0 += 32)
      acc = wmma_bf16(load_a16(arow_h + k0, a_kh),
                      load_b16(s_whh, (uint32_t)(role * JW + b_n) * RNN + k0 + b_kh), acc);

    // D layout: VGPR v -> M = v + 8*(lane>=16), N = lane&15
    {
      const int n  = lane & 15;
      const int mb = mtile * 16 + ((lane >> 4) << 3);
      if (role == 0) {
#pragma unroll
        for (int v = 0; v < 8; ++v) s_acc[0][mb + v][n] = acc[v];
      } else if (role == 1) {
#pragma unroll
        for (int v = 0; v < 8; ++v) s_acc[1][mb + v][n] = acc[v];
      } else {
#pragma unroll
        for (int v = 0; v < 8; ++v) { s_acc[3][mb + v][n] = acc[v]; s_acc[2][mb + v][n] = acc2[v]; }
      }
    }
    __syncthreads();

    // elementwise gates + state update for this WG's 64x16 slice
    const float* hr = hf + (size_t)p * (BATCH * RNN);
    float*  hw  = hf + (size_t)(1 - p) * (BATCH * RNN);
    __bf16* hbw = hb + (size_t)(1 - p) * (BATCH * RNN);
    for (int i = tid; i < BATCH * JW; i += THREADS) {
      const int b = i >> 4, jj = i & 15, j = j0 + jj;
      const float rr = sigmoidf(s_acc[0][b][jj] + bih[j] + bhh[j]);
      const float zz = sigmoidf(s_acc[1][b][jj] + bih[RNN + j] + bhh[RNN + j]);
      const float nn = tanhf(s_acc[2][b][jj] + bih[2 * RNN + j] +
                             rr * (s_acc[3][b][jj] + bhh[2 * RNN + j]));
      const float hp = hr[b * RNN + j];
      const float hy = zz * nn + (1.0f - zz) * hp;
      hw[b * RNN + j]  = hy;
      hbw[b * RNN + j] = (__bf16)hy;
    }

    // grid-wide sync (all WGs must finish step t before step t+1 reads h)
    __threadfence();
    __syncthreads();
    if (tid == 0) {
      __hip_atomic_fetch_add(ctr, 1u, __ATOMIC_RELEASE, __HIP_MEMORY_SCOPE_AGENT);
      const unsigned target = (unsigned)(t + 1) * (unsigned)NWG;
      while (__hip_atomic_load(ctr, __ATOMIC_ACQUIRE, __HIP_MEMORY_SCOPE_AGENT) < target)
        __builtin_amdgcn_s_sleep(2);
    }
    __syncthreads();
    __threadfence();
  }
}

// final FC: out[b][o] = fc_b[o] + h_last[b] . fc_w[o]
__global__ void k_out(const float* __restrict__ hfinal, const float* __restrict__ fcw,
                      const float* __restrict__ fcb, float* __restrict__ out) {
  const int tid = threadIdx.x;              // 256 = 64*4
  const int b = tid >> 2, o = tid & 3;
  float acc = fcb[o];
  const float* h = hfinal + (size_t)b * RNN;
  const float* w = fcw + (size_t)o * RNN;
  for (int k = 0; k < RNN; ++k) acc += h[k] * w[k];
  out[b * OUTN + o] = acc;
}

extern "C" void kernel_launch(void* const* d_in, const int* in_sizes, int n_in,
                              void* d_out, int out_size, void* d_ws, size_t ws_size,
                              hipStream_t stream) {
  (void)in_sizes; (void)n_in; (void)out_size; (void)ws_size;
  const int*   x   = (const int*)d_in[0];
  const float* tab = (const float*)d_in[1];
  const float* wih = (const float*)d_in[2];
  const float* whh = (const float*)d_in[3];
  const float* bih = (const float*)d_in[4];
  const float* bhh = (const float*)d_in[5];
  const float* fcw = (const float*)d_in[6];
  const float* fcb = (const float*)d_in[7];
  float* out = (float*)d_out;

  char* ws = (char*)d_ws;
  const size_t o_wih = 0;
  const size_t o_whh = o_wih + (size_t)3 * RNN * EMB * 2;          //  3.0 MB
  const size_t o_emb = o_whh + (size_t)3 * RNN * RNN * 2;          //  6.0 MB
  const size_t o_hf  = o_emb + (size_t)T_STEPS * BATCH * EMB * 2;  // 32.0 MB
  const size_t o_hb  = o_hf + (size_t)2 * BATCH * RNN * 4;
  const size_t o_ctr = o_hb + (size_t)2 * BATCH * RNN * 2;
  __bf16*   ws_wih = (__bf16*)(ws + o_wih);
  __bf16*   ws_whh = (__bf16*)(ws + o_whh);
  __bf16*   ws_emb = (__bf16*)(ws + o_emb);
  float*    ws_hf  = (float*)(ws + o_hf);
  __bf16*   ws_hb  = (__bf16*)(ws + o_hb);
  unsigned* ws_ctr = (unsigned*)(ws + o_ctr);

  const int n1 = 3 * RNN * EMB, n2 = 3 * RNN * RNN;
  k_cvt<<<(n1 + 255) / 256, 256, 0, stream>>>(wih, ws_wih, n1);
  k_cvt<<<(n2 + 255) / 256, 256, 0, stream>>>(whh, ws_whh, n2);
  k_embed<<<T_STEPS * BATCH, 128, 0, stream>>>(x, tab, ws_emb);
  k_init<<<256, 256, 0, stream>>>(ws_hf, ws_hb, ws_ctr);
  k_gru<<<NWG, THREADS, 0, stream>>>(ws_wih, ws_whh, ws_emb, bih, bhh,
                                     ws_hf, ws_hb, ws_ctr);
  // after 512 steps the final h is in buffer 0 (last write goes to 1-(511&1) = 0)
  k_out<<<1, BATCH * OUTN, 0, stream>>>(ws_hf, fcw, fcb, out);
}
